// OTFlow_40587440947279
// MI455X (gfx1250) — matrix operand order repaired and checked by reading, not compiled, not run on hardware
//
#include <hip/hip_runtime.h>

// ---------------------------------------------------------------------------
// OT-Flow RK4 integrator for MI455X (gfx1250), wave32 + v_wmma_f32_16x16x32_bf16.
// Each wave integrates 16 samples entirely on-chip. All weights (incl. the
// precomputed W[j, d*64+i] = K1[i,j]*K0[j,d] matrix that linearizes the
// dominant einsum) live in LDS. 154 WMMA ops per odefun eval per wave.
// Round 4: KJ loop kept rolled (#pragma unroll 1) to bound register liveness
// below 256 and eliminate scratch spills / s_set_vgpr_msb traffic.
// ---------------------------------------------------------------------------

typedef __bf16 bf16_t;
typedef __attribute__((ext_vector_type(16))) __bf16 v16bf;
typedef __attribute__((ext_vector_type(8)))  float  v8f;

#define DEV static __device__ __forceinline__

union FragU { v16bf v; uint4 q[2]; };

// A-fragment (16x32 bf16, ISA 7.12.2 layout) from LDS row-major [16][64].
DEV v16bf loadA_frag(const bf16_t* base, int lane, int k0) {
  const int m = lane & 15, hi = lane >> 4;
  const bf16_t* p = base + m * 64 + k0 + hi * 8;
  FragU f;
  f.q[0] = *(const uint4*)(p);
  f.q[1] = *(const uint4*)(p + 16);
  return f.v;
}

// B-fragment (32x16 bf16) from LDS stored column-major (col's K contiguous).
DEV v16bf loadB_frag(const bf16_t* base, int colBlock, int k0, int kstride, int lane) {
  const int ncol = colBlock + (lane & 15), hi = lane >> 4;
  const bf16_t* p = base + ncol * kstride + k0 + hi * 8;
  FragU f;
  f.q[0] = *(const uint4*)(p);
  f.q[1] = *(const uint4*)(p + 16);
  return f.v;
}

DEV v8f splat8(float x) {
  v8f v;
  #pragma unroll
  for (int i = 0; i < 8; ++i) v[i] = x;
  return v;
}

DEV v8f wmma_bf(v16bf a, v16bf b, v8f c) {
  return __builtin_amdgcn_wmma_f32_16x16x32_bf16(false, a, false, b, (short)0, c,
                                                 false, false);
}

DEV float setsign(float mag, float s) {   // mag >= 0; copy sign bit of s
  return __uint_as_float(__float_as_uint(mag) | (__float_as_uint(s) & 0x80000000u));
}
// raw hardware transcendentals: v_exp_f32 = exp2, v_log_f32 = log2.
DEV float exp_n2(float ax) {              // exp(-2*ax), ax >= 0 -> result in (0,1]
  return __builtin_amdgcn_exp2f(ax * -2.8853900817779268f);  // -2*log2(e)
}
DEV float log_e(float x) {                // ln(x), x in (1,2] -> no fixups needed
  return __builtin_amdgcn_logf(x) * 0.6931471805599453f;     // log2(x)*ln2
}
DEV float tanh_fast(float x) {
  float e = exp_n2(fabsf(x));
  float r = __builtin_amdgcn_rcpf(1.0f + e);
  return setsign((1.0f - e) * r, x);
}
DEV float red16(float v) {                // sum across each 16-lane half of wave32
  v += __shfl_xor(v, 1, 16);
  v += __shfl_xor(v, 2, 16);
  v += __shfl_xor(v, 4, 16);
  v += __shfl_xor(v, 8, 16);
  return v;
}

struct Shm {
  const bf16_t *W;    // [1024 cols][64 K]  W[col=d*64+i][j] = K1[i,j]*K0[j,d]
  const bf16_t *K1T;  // [64 N][64 K]  B for u0 @ K1^T
  const bf16_t *K1N;  // [64 N][64 K]  B for a2 @ K1
  const bf16_t *K0T;  // [64 N][32 K]  B for s @ K0^T   (K padded 17->32)
  const bf16_t *K0N;  // [32 N][64 K]  B for tz @ K0    (N padded 17->32)
  const bf16_t *SA;   // [32 N][32 K]  symA = A^T A padded
};

struct LaneConst {    // per-lane (m15) feature-column constants, resident in regs
  float b0c[4], b1c[4], wc[4], sqc[4];
  float bp0, bp1, sumA2;
};

// One odefun evaluation for this wave's 16 samples (single inlined copy).
DEV void odefun_eval(const Shm& S, const LaneConst& L,
                     const float* __restrict__ zA, float t,
                     float* __restrict__ dz, bf16_t* aSt, bf16_t* toSt, int lane)
{
  const int m15 = lane & 15, hi = lane >> 4;

  // ---- stage s = [x, t, 0...] as 16x32 bf16 rows (row stride 64) ----
  {
    float pad = (lane == 16) ? t : 0.0f;
    #pragma unroll
    for (int r = 0; r < 16; ++r) {
      float sv = (lane < 16) ? zA[r * 19 + lane] : pad;
      aSt[r * 64 + lane] = (bf16_t)sv;
    }
  }
  v16bf sF = loadA_frag(aSt, lane, 0);

  // ---- opening = s @ K0^T + b0; consume tile immediately (nothing kept) ----
  #pragma unroll
  for (int n4 = 0; n4 < 4; ++n4) {
    v8f acc = splat8(L.b0c[n4]);
    acc = wmma_bf(sF, loadB_frag(S.K0T, n4 * 16, 0, 32, lane), acc);
    #pragma unroll
    for (int r = 0; r < 8; ++r) {
      float xo = acc[r];
      float ax = fabsf(xo);
      float e  = exp_n2(ax);                                // shared exp
      float th = setsign((1.0f - e) * __builtin_amdgcn_rcpf(1.0f + e), xo);
      float u0 = ax + log_e(1.0f + e);                      // antideriv_tanh
      aSt [(r + hi * 8) * 64 + n4 * 16 + m15] = (bf16_t)u0;
      toSt[(r + hi * 8) * 64 + n4 * 16 + m15] = (bf16_t)th;
    }
  }
  v16bf uF0 = loadA_frag(aSt, lane, 0);
  v16bf uF1 = loadA_frag(aSt, lane, 32);

  // ---- feat1 = u0 @ K1^T + b1; keep only cT; stage a2 = tanh(feat1)*w ----
  float cT[4][8];
  #pragma unroll
  for (int n4 = 0; n4 < 4; ++n4) {
    v8f acc = splat8(L.b1c[n4]);
    acc = wmma_bf(uF0, loadB_frag(S.K1T, n4 * 16, 0, 64, lane), acc);
    acc = wmma_bf(uF1, loadB_frag(S.K1T, n4 * 16, 32, 64, lane), acc);
    float ws = L.wc[n4];
    #pragma unroll
    for (int r = 0; r < 8; ++r) {
      float tf = tanh_fast(acc[r]);
      cT[n4][r] = (1.0f - tf * tf) * ws;
      aSt[(r + hi * 8) * 64 + n4 * 16 + m15] = (bf16_t)(tf * ws);
    }
  }
  v16bf aF0 = loadA_frag(aSt, lane, 0);
  v16bf aF1 = loadA_frag(aSt, lane, 32);

  // ---- z1 = w + a2 @ K1; trH0 partial; stage tz = tanhopen * z1.
  //      'to' re-read from LDS (bf16) instead of living in VGPRs. ----
  float trAcc[8];
  #pragma unroll
  for (int r = 0; r < 8; ++r) trAcc[r] = 0.0f;
  #pragma unroll
  for (int n4 = 0; n4 < 4; ++n4) {
    v8f acc = splat8(L.wc[n4]);
    acc = wmma_bf(aF0, loadB_frag(S.K1N, n4 * 16, 0, 64, lane), acc);
    acc = wmma_bf(aF1, loadB_frag(S.K1N, n4 * 16, 32, 64, lane), acc);
    float sq = L.sqc[n4];
    #pragma unroll
    for (int r = 0; r < 8; ++r) {
      float to_ = (float)toSt[(r + hi * 8) * 64 + n4 * 16 + m15];
      float zz  = acc[r];
      trAcc[r] += (1.0f - to_ * to_) * zz * sq;
      aSt[(r + hi * 8) * 64 + n4 * 16 + m15] = (bf16_t)(to_ * zz);
    }
  }

  // ---- KJ = tanhopen @ W with fused c-weighted square-accumulate.
  //      Kept as a REAL loop (8 WMMA/iter) to bound register liveness. ----
  v16bf toF0 = loadA_frag(toSt, lane, 0);
  v16bf toF1 = loadA_frag(toSt, lane, 32);
  float qAcc[8];
  #pragma unroll
  for (int r = 0; r < 8; ++r) qAcc[r] = 0.0f;
  #pragma unroll 1
  for (int d = 0; d < 16; ++d) {
    #pragma unroll
    for (int ig = 0; ig < 4; ++ig) {
      const int cb = d * 64 + ig * 16;
      v8f kj = splat8(0.0f);
      kj = wmma_bf(toF0, loadB_frag(S.W, cb, 0, 64, lane), kj);
      kj = wmma_bf(toF1, loadB_frag(S.W, cb, 32, 64, lane), kj);
      #pragma unroll
      for (int r = 0; r < 8; ++r) { float v = kj[r]; qAcc[r] += cT[ig][r] * v * v; }
    }
  }

  // ---- grad = tz @ K0 + s @ symA + b  (6 WMMA), after KJ to cut pressure ----
  v16bf tzF0 = loadA_frag(aSt, lane, 0);
  v16bf tzF1 = loadA_frag(aSt, lane, 32);
  v8f g0 = splat8(L.bp0);
  g0 = wmma_bf(tzF0, loadB_frag(S.K0N, 0, 0, 64, lane), g0);
  g0 = wmma_bf(tzF1, loadB_frag(S.K0N, 0, 32, 64, lane), g0);
  g0 = wmma_bf(sF,   loadB_frag(S.SA,  0, 0, 32, lane), g0);
  v8f g1 = splat8(L.bp1);
  g1 = wmma_bf(tzF0, loadB_frag(S.K0N, 16, 0, 64, lane), g1);
  g1 = wmma_bf(tzF1, loadB_frag(S.K0N, 16, 32, 64, lane), g1);
  g1 = wmma_bf(sF,   loadB_frag(S.SA,  16, 0, 32, lane), g1);

  // ---- reductions + dz assembly (ALPHA0 = H_RES = 1) ----
  #pragma unroll
  for (int r = 0; r < 8; ++r) {
    float trH = red16(trAcc[r] + qAcc[r]) + L.sumA2;
    float dv  = 0.5f * red16(g0[r] * g0[r]);   // dx = -grad -> dx^2 = grad^2
    int row = r + hi * 8;
    dz[row * 19 + m15] = -g0[r];               // dx (cols 0..15)
    if (m15 == 0) {
      dz[row * 19 + 16] = -trH;                // dl
      dz[row * 19 + 17] = dv;                  // dv
      dz[row * 19 + 18] = fabsf(dv - g1[r]);   // dr = |-grad_t + dv|
    }
  }
}

// ---------------------------------------------------------------------------
constexpr int WAVES = 8;
constexpr int TPB   = WAVES * 32;

constexpr int OFF_WB   = 0;
constexpr int OFF_K1T  = OFF_WB  + 1024 * 64 * 2;   // 131072
constexpr int OFF_K1N  = OFF_K1T + 64 * 64 * 2;
constexpr int OFF_K0T  = OFF_K1N + 64 * 64 * 2;
constexpr int OFF_K0N  = OFF_K0T + 64 * 32 * 2;
constexpr int OFF_SA   = OFF_K0N + 32 * 64 * 2;
constexpr int OFF_B0   = OFF_SA  + 32 * 32 * 2;
constexpr int OFF_B1   = OFF_B0  + 64 * 4;
constexpr int OFF_WV   = OFF_B1  + 64 * 4;
constexpr int OFF_SQ   = OFF_WV  + 64 * 4;
constexpr int OFF_BP   = OFF_SQ  + 64 * 4;
constexpr int OFF_SA2  = OFF_BP  + 32 * 4;
constexpr int OFF_WAVE = OFF_SA2 + 16;                       // 158864 (16B aligned)
constexpr int WAVE_BYTES = 16*64*2 + 16*64*2 + 4 * (304*4);  // 8960
constexpr int SMEM_TOTAL = OFF_WAVE + WAVES * WAVE_BYTES;    // 230544 < 320KB/WGP

__global__ __launch_bounds__(TPB, 1)
void otflow_rk4_wmma(const float* __restrict__ x,  const float* __restrict__ wv,
                     const float* __restrict__ A,  const float* __restrict__ bvec,
                     const float* __restrict__ K0, const float* __restrict__ b0,
                     const float* __restrict__ K1, const float* __restrict__ b1,
                     const int* __restrict__ ntp,  float* __restrict__ out, int n)
{
  extern __shared__ char smem[];
  bf16_t* W_B = (bf16_t*)(smem + OFF_WB);
  bf16_t* K1T = (bf16_t*)(smem + OFF_K1T);
  bf16_t* K1N = (bf16_t*)(smem + OFF_K1N);
  bf16_t* K0T = (bf16_t*)(smem + OFF_K0T);
  bf16_t* K0N = (bf16_t*)(smem + OFF_K0N);
  bf16_t* SAB = (bf16_t*)(smem + OFF_SA);
  float*  b0f = (float*)(smem + OFF_B0);
  float*  b1f = (float*)(smem + OFF_B1);
  float*  wf  = (float*)(smem + OFF_WV);
  float*  sqf = (float*)(smem + OFF_SQ);
  float*  bpf = (float*)(smem + OFF_BP);
  float*  sA2 = (float*)(smem + OFF_SA2);

  const int tid = threadIdx.x;

  // ---- cooperative weight prep (once per block) ----
  for (int e = tid; e < 1024 * 64; e += TPB) {
    int col = e >> 6, j = e & 63;
    int d = col >> 6, i = col & 63;
    W_B[e] = (bf16_t)(K1[i * 64 + j] * K0[j * 17 + d]);
  }
  for (int e = tid; e < 64 * 64; e += TPB) {
    int j = e >> 6, k = e & 63;
    K1T[e] = (bf16_t)K1[j * 64 + k];
    K1N[e] = (bf16_t)K1[k * 64 + j];
  }
  for (int e = tid; e < 64 * 32; e += TPB) {
    int j = e >> 5, k = e & 31;
    K0T[e] = (bf16_t)((k < 17) ? K0[j * 17 + k] : 0.0f);
  }
  for (int e = tid; e < 32 * 64; e += TPB) {
    int j = e >> 6, k = e & 63;
    K0N[e] = (bf16_t)((j < 17) ? K0[k * 17 + j] : 0.0f);
  }
  for (int e = tid; e < 32 * 32; e += TPB) {
    int j = e >> 5, k = e & 31;
    float s = 0.0f;
    if (j < 17 && k < 17)
      for (int r = 0; r < 10; ++r) s += A[r * 17 + k] * A[r * 17 + j];
    SAB[e] = (bf16_t)s;
  }
  for (int e = tid; e < 64; e += TPB) {
    b0f[e] = b0[e]; b1f[e] = b1[e]; wf[e] = wv[e];
    float s = 0.0f;
    for (int d = 0; d < 16; ++d) { float v = K0[e * 17 + d]; s += v * v; }
    sqf[e] = s;
  }
  for (int e = tid; e < 32; e += TPB) bpf[e] = (e < 17) ? bvec[e] : 0.0f;
  if (tid == 0) {
    float s = 0.0f;
    for (int r = 0; r < 10; ++r)
      for (int c = 0; c < 16; ++c) { float v = A[r * 17 + c]; s += v * v; }
    sA2[0] = s;
  }
  __syncthreads();

  Shm S { W_B, K1T, K1N, K0T, K0N, SAB };

  const int wave = tid >> 5, lane = tid & 31;
  const int m15 = lane & 15;

  // per-lane scalar weight columns, resident in VGPRs for the whole kernel
  LaneConst L;
  #pragma unroll
  for (int n4 = 0; n4 < 4; ++n4) {
    L.b0c[n4] = b0f[n4 * 16 + m15];
    L.b1c[n4] = b1f[n4 * 16 + m15];
    L.wc [n4] = wf [n4 * 16 + m15];
    L.sqc[n4] = sqf[n4 * 16 + m15];
  }
  L.bp0 = bpf[m15];
  L.bp1 = bpf[16 + m15];
  L.sumA2 = sA2[0];

  char* wb = smem + OFF_WAVE + wave * WAVE_BYTES;
  bf16_t* aSt  = (bf16_t*)wb;
  bf16_t* toSt = (bf16_t*)(wb + 2048);
  float* zCur  = (float*)(wb + 4096);
  float* zNxt  = zCur + 304;
  float* zArg  = zNxt + 304;
  float* dzB   = zArg + 304;

  const int base = (blockIdx.x * WAVES + wave) * 16;

  for (int e = lane; e < 304; e += 32) {
    int r = e / 19, c = e - r * 19;
    zCur[e] = (c < 16) ? x[(size_t)(base + r) * 16 + c] : 0.0f;
  }

  const int nt = ntp[0];
  const float hstep = 1.0f / (float)nt;

  for (int k = 0; k < nt; ++k) {
    const float t0 = k * hstep;
    for (int e = lane; e < 304; e += 32) { float z = zCur[e]; zNxt[e] = z; zArg[e] = z; }

    // RK4 as a rolled 4-stage loop: exactly ONE inlined copy of odefun_eval.
    #pragma unroll 1
    for (int st = 0; st < 4; ++st) {
      const float tS = t0 + ((st == 0) ? 0.0f : (st == 3) ? hstep : 0.5f * hstep);
      odefun_eval(S, L, zArg, tS, dzB, aSt, toSt, lane);
      const float bc = (st == 0 || st == 3) ? (1.0f / 6.0f) : (1.0f / 3.0f);
      const float an = (st <= 1) ? 0.5f : (st == 2) ? 1.0f : 0.0f;  // a-coef of NEXT stage
      for (int e = lane; e < 304; e += 32) {
        float kv = hstep * dzB[e];
        zNxt[e] += bc * kv;
        zArg[e]  = zCur[e] + an * kv;
      }
    }
    for (int e = lane; e < 304; e += 32) zCur[e] = zNxt[e];
  }

  // ---- output: [n*16] x, [n] l, [n] v, [n] r ----
  for (int e = lane; e < 256; e += 32) {
    int r = e >> 4, c = e & 15;
    out[(size_t)(base + r) * 16 + c] = zCur[r * 19 + c];
  }
  if (lane < 16) {
    out[(size_t)n * 16 + (base + lane)] = zCur[lane * 19 + 16];
    out[(size_t)n * 17 + (base + lane)] = zCur[lane * 19 + 17];
    out[(size_t)n * 18 + (base + lane)] = zCur[lane * 19 + 18];
  }
}

extern "C" void kernel_launch(void* const* d_in, const int* in_sizes, int n_in,
                              void* d_out, int out_size, void* d_ws, size_t ws_size,
                              hipStream_t stream) {
  const float* x  = (const float*)d_in[0];
  const float* w  = (const float*)d_in[1];
  const float* A  = (const float*)d_in[2];
  const float* b  = (const float*)d_in[3];
  const float* K0 = (const float*)d_in[4];
  const float* b0 = (const float*)d_in[5];
  const float* K1 = (const float*)d_in[6];
  const float* b1 = (const float*)d_in[7];
  const int*   nt = (const int*)d_in[8];
  float* out = (float*)d_out;
  (void)d_ws; (void)ws_size; (void)n_in; (void)out_size;

  const int n = in_sizes[0] / 16;          // 32768
  const int blocks = n / (WAVES * 16);     // 256
  otflow_rk4_wmma<<<blocks, TPB, SMEM_TOTAL, stream>>>(x, w, A, b, K0, b0, K1, b1,
                                                       nt, out, n);
}
